// x_convs_74019466379459
// MI455X (gfx1250) — matrix operand order repaired and checked by reading, not compiled, not run on hardware
//
#include <hip/hip_runtime.h>
#include <hip/hip_bf16.h>

typedef __attribute__((ext_vector_type(16))) _Float16 v16h;
typedef __attribute__((ext_vector_type(8)))  _Float16 v8h;
typedef __attribute__((ext_vector_type(8)))  float    v8f;
typedef __attribute__((ext_vector_type(4)))  float    v4f;

#define EPSV 1e-5f

__device__ __forceinline__ float eluf(float v) {
    return v > 0.f ? v : (__expf(v) - 1.f);
}
__device__ __forceinline__ float sigf(float v) {
    return 1.f / (1.f + __expf(-v));
}
__device__ __forceinline__ v16h cat8(v8h lo, v8h hi) {
    return __builtin_shufflevector(lo, hi, 0, 1, 2, 3, 4, 5, 6, 7,
                                   8, 9, 10, 11, 12, 13, 14, 15);
}

// ---------------------------------------------------------------------------
// prep: transpose+convert conv weights into WMMA-friendly layouts.
//  w2t f16: [branch][o][r'] with r' = kk*192 + c   (2*48*1728 elements)
//  w3t f32: [branch][kk][c][o]                     (2*9*96*4 elements)
// ---------------------------------------------------------------------------
__global__ void prep_kernel(const float* __restrict__ w2a, const float* __restrict__ w2b,
                            const float* __restrict__ w3a, const float* __restrict__ w3b,
                            _Float16* __restrict__ w2t, float* __restrict__ w3t)
{
    int idx = blockIdx.x * 256 + threadIdx.x;
    if (idx < 165888) {                       // w2 transpose
        int branch = idx / 82944;
        int i = idx - branch * 82944;
        int o = i / 1728;
        int r = i - o * 1728;
        int kk = r / 192;
        int c  = r - kk * 192;
        const float* src = branch ? w2b : w2a;
        w2t[idx] = (_Float16)src[o * 1728 + c * 9 + kk];
    } else if (idx < 172800) {                // w3 transpose
        int j = idx - 165888;
        int branch = j / 3456;
        int t = j - branch * 3456;
        int kk = t / 384;
        int rem = t - kk * 384;
        int c = rem >> 2, o = rem & 3;
        const float* src = branch ? w3b : w3a;
        w3t[j] = src[(o * 96 + c) * 9 + kk];
    }
}

// ---------------------------------------------------------------------------
// conv1: x4 (4ch) -> masked 3x3 conv -> 96ch -> pono -> concat_elu
// h1 layout (channel-fastest): [branch][b][pos][192] f16
// ---------------------------------------------------------------------------
__global__ void conv1_kernel(const float* __restrict__ x,
                             const float* __restrict__ m1, const float* __restrict__ m2,
                             const float* __restrict__ w_a, const float* __restrict__ b_a,
                             const float* __restrict__ w_b, const float* __restrict__ b_b,
                             _Float16* __restrict__ h1)
{
    __shared__ float sv[64 * 96];
    __shared__ float sw[96 * 36 + 96];
    int tid    = blockIdx.x * 64 + threadIdx.x;      // 0..65535
    int branch = tid >> 15;
    int pos    = tid & 32767;
    int b = pos >> 12, l = pos & 4095;
    int y = l >> 6, xc0 = l & 63;
    const float* m    = branch ? m2  : m1;
    const float* w    = branch ? w_b : w_a;
    const float* bias = branch ? b_b : b_a;

    for (int i = threadIdx.x; i < 96 * 36 + 96; i += 64)
        sw[i] = (i < 3456) ? w[i] : bias[i - 3456];
    __syncthreads();

    float pm[36];
#pragma unroll
    for (int kk = 0; kk < 9; ++kk) {
        int t3 = kk / 3;
        int yy = y + t3 - 1, xx = xc0 + (kk - t3 * 3) - 1;
        bool in = ((unsigned)yy < 64u) && ((unsigned)xx < 64u);
        float mk = m[kk * 4096 + l];
#pragma unroll
        for (int c = 0; c < 4; ++c) {
            float v = 0.f;
            if (in) v = (c < 3) ? x[((b * 3 + c) << 12) + (yy << 6) + xx] : 1.0f;
            pm[c * 9 + kk] = v * mk;
        }
    }
    float s1 = 0.f, s2 = 0.f;
    for (int o = 0; o < 96; ++o) {
        float v = sw[3456 + o];
#pragma unroll
        for (int i = 0; i < 36; ++i) v += sw[o * 36 + i] * pm[i];
        sv[o * 64 + threadIdx.x] = v;
        s1 += v; s2 += v * v;
    }
    float mean = s1 * (1.f / 96.f);
    float var  = s2 * (1.f / 96.f) - mean * mean;
    float inv  = rsqrtf(var + EPSV);
    _Float16* outp = h1 + (size_t)pos * 192 + (size_t)branch * (32768ull * 192);
    for (int o = 0; o < 96; ++o) {
        float t = (sv[o * 64 + threadIdx.x] - mean) * inv;
        outp[o]      = (_Float16)eluf(t);
        outp[96 + o] = (_Float16)eluf(-t);
    }
}

// ---------------------------------------------------------------------------
// conv2: 192ch masked 3x3 conv -> 48ch via WMMA implicit GEMM.
// Contraction order r' = kk*192 + c: kk constant per 32-chunk, channels
// contiguous -> b128 fragment loads. One wave per 16-position tile.
// h2 layout (channel-fastest): [branch][b][pos][96] f16
// ---------------------------------------------------------------------------
__global__ void conv2_kernel(const _Float16* __restrict__ h1,
                             const float* __restrict__ m1, const float* __restrict__ m2,
                             const _Float16* __restrict__ w2t,
                             const float* __restrict__ bias_a, const float* __restrict__ bias_b,
                             _Float16* __restrict__ h2)
{
    __shared__ float sc[48 * 16];
    int lane   = threadIdx.x;           // wave32, one wave per block
    int blk    = blockIdx.x;
    int branch = blk >> 11;
    int tile   = blk & 2047;
    int base   = tile << 4;             // 0..32767 position base
    int b  = base >> 12;
    int l0 = base & 4095;
    const float* m    = branch ? m2     : m1;
    const float* bias = branch ? bias_b : bias_a;
    const _Float16* hin = h1 + ((size_t)branch * 32768ull + (size_t)(b << 12)) * 192;
    const _Float16* wt  = w2t + (size_t)branch * 82944;

    bool hi = lane >= 16;
    int  n  = lane & 15;
    int  l  = l0 + n;
    int  y  = l >> 6, xc0 = l & 63;
    int  ar = lane & 15;

    v8f acc0 = {}, acc1 = {}, acc2 = {};
#pragma unroll
    for (int kk = 0; kk < 9; ++kk) {
        int t3 = kk / 3;
        int yy = y + t3 - 1, xx = xc0 + (kk - t3 * 3) - 1;
        bool in = ((unsigned)yy < 64u) && ((unsigned)xx < 64u);
        int poff = in ? ((yy << 6) + xx) * 192 : 0;
        _Float16 hm = (_Float16)(in ? m[kk * 4096 + l] : 0.f);
        const _Float16* hrow = hin + poff;
#pragma unroll
        for (int sub = 0; sub < 6; ++sub) {
            int cb = sub * 32 + (hi ? 8 : 0);
            v8h blo = *(const v8h*)(hrow + cb);
            v8h bhi = *(const v8h*)(hrow + cb + 16);
            v16h hv = cat8(blo, bhi);
            v16h bf;
#pragma unroll
            for (int j = 0; j < 16; ++j) bf[j] = hv[j] * hm;
            int ab = kk * 192 + sub * 32 + (hi ? 8 : 0);
            {
                const _Float16* wr = wt + (size_t)ar * 1728 + ab;
                v16h af = cat8(*(const v8h*)wr, *(const v8h*)(wr + 16));
                acc0 = __builtin_amdgcn_wmma_f32_16x16x32_f16(false, af, false, bf, (short)0, acc0, false, false);
            }
            {
                const _Float16* wr = wt + (size_t)(16 + ar) * 1728 + ab;
                v16h af = cat8(*(const v8h*)wr, *(const v8h*)(wr + 16));
                acc1 = __builtin_amdgcn_wmma_f32_16x16x32_f16(false, af, false, bf, (short)0, acc1, false, false);
            }
            {
                const _Float16* wr = wt + (size_t)(32 + ar) * 1728 + ab;
                v16h af = cat8(*(const v8h*)wr, *(const v8h*)(wr + 16));
                acc2 = __builtin_amdgcn_wmma_f32_16x16x32_f16(false, af, false, bf, (short)0, acc2, false, false);
            }
        }
    }
    // transpose accumulators through LDS for per-position pono
    int moff = hi ? 8 : 0;
#pragma unroll
    for (int r = 0; r < 8; ++r) {
        sc[(r + moff) * 16 + n]      = acc0[r];
        sc[(16 + r + moff) * 16 + n] = acc1[r];
        sc[(32 + r + moff) * 16 + n] = acc2[r];
    }
    __syncthreads();
    if (lane < 16) {
        float vals[48];
        float s1 = 0.f;
#pragma unroll
        for (int o = 0; o < 48; ++o) {
            float v = sc[o * 16 + lane] + bias[o];
            vals[o] = v; s1 += v;
        }
        float mean = s1 * (1.f / 48.f);
        float s2 = 0.f;
#pragma unroll
        for (int o = 0; o < 48; ++o) { float d = vals[o] - mean; s2 += d * d; }
        float inv = rsqrtf(s2 * (1.f / 48.f) + EPSV);
        int pos = (b << 12) + l0 + lane;
        _Float16* outp = h2 + ((size_t)branch * 32768ull + pos) * 96;
#pragma unroll
        for (int o = 0; o < 48; ++o) {
            float t = (vals[o] - mean) * inv;
            outp[o]      = (_Float16)eluf(t);
            outp[48 + o] = (_Float16)eluf(-t);
        }
    }
}

// ---------------------------------------------------------------------------
// conv3 (dilation 2): 96ch -> 4ch scalar; vectorized activation/weight loads.
// h3 layout: [branch][b][8][4096] f32
// ---------------------------------------------------------------------------
__global__ void conv3_kernel(const _Float16* __restrict__ h2,
                             const float* __restrict__ m1, const float* __restrict__ m2,
                             const float* __restrict__ w3t,
                             const float* __restrict__ b_a, const float* __restrict__ b_b,
                             float* __restrict__ h3)
{
    int tid    = blockIdx.x * 256 + threadIdx.x;
    int branch = tid >> 15;
    int pos    = tid & 32767;
    int b = pos >> 12, l = pos & 4095;
    int y = l >> 6, xc0 = l & 63;
    const float* m    = branch ? m2  : m1;
    const float* bias = branch ? b_b : b_a;
    const _Float16* hin = h2 + ((size_t)branch * 32768ull + (size_t)(b << 12)) * 96;
    const float* wtb = w3t + branch * 3456;

    v4f acc = { bias[0], bias[1], bias[2], bias[3] };
#pragma unroll
    for (int kk = 0; kk < 9; ++kk) {
        int t3 = kk / 3;
        int yy = y + (t3 - 1) * 2, xx = xc0 + ((kk - t3 * 3) - 1) * 2;
        bool in = ((unsigned)yy < 64u) && ((unsigned)xx < 64u);
        if (!in) continue;
        float mk = m[kk * 4096 + l];
        const _Float16* hrow = hin + ((yy << 6) + xx) * 96;
        const v4f* wr = (const v4f*)(wtb + kk * 384);
#pragma unroll
        for (int cb = 0; cb < 96; cb += 8) {
            v8h hv = *(const v8h*)(hrow + cb);
#pragma unroll
            for (int e = 0; e < 8; ++e) {
                float v = (float)hv[e] * mk;
                acc += wr[cb + e] * v;
            }
        }
    }
    float mean = (acc[0] + acc[1] + acc[2] + acc[3]) * 0.25f;
    float s2 = 0.f;
#pragma unroll
    for (int o = 0; o < 4; ++o) { float d = acc[o] - mean; s2 += d * d; }
    float inv = rsqrtf(s2 * 0.25f + EPSV);
    size_t obase = (((size_t)branch * 8 + b) * 8) << 12;
#pragma unroll
    for (int o = 0; o < 4; ++o) {
        float t = (acc[o] - mean) * inv;
        h3[obase + ((size_t)o << 12) + l]       = eluf(t);
        h3[obase + ((size_t)(o + 4) << 12) + l] = eluf(-t);
    }
}

// ---------------------------------------------------------------------------
// combine: x_conv = h3a*h3b; pono(a); gated residual; padded flat [16][16384]
// ---------------------------------------------------------------------------
__global__ void combine_kernel(const float* __restrict__ x,
                               const float* __restrict__ h3,
                               float* __restrict__ flat)
{
    int idx = blockIdx.x * 256 + threadIdx.x;   // 0..262143
    int row = idx >> 14;
    int col = idx & 16383;
    float val = 0.f;
    if (row < 8) {
        int b = row, c = col >> 12, l = col & 4095;
        const float* ha = h3 + (((size_t)b * 8) << 12);
        const float* hb = h3 + (((size_t)(8 + b) * 8) << 12);
        float av[4];
#pragma unroll
        for (int i = 0; i < 4; ++i)
            av[i] = ha[((size_t)i << 12) + l] * hb[((size_t)i << 12) + l];
        float g = ha[((size_t)(4 + c) << 12) + l] * hb[((size_t)(4 + c) << 12) + l];
        float mean = (av[0] + av[1] + av[2] + av[3]) * 0.25f;
        float s2 = 0.f;
#pragma unroll
        for (int i = 0; i < 4; ++i) { float d = av[i] - mean; s2 += d * d; }
        float inv = rsqrtf(s2 * 0.25f + EPSV);
        float an  = (av[c] - mean) * inv;
        float x4v = (c < 3) ? x[((b * 3 + c) << 12) + l] : 1.0f;
        val = x4v + an * sigf(g);
    }
    flat[idx] = val;
}

// ---------------------------------------------------------------------------
// linear layers via WMMA, 8 waves per block split K, LDS reduce.
// mode 0: write padded h4 [16][4096]; mode 1: write both output copies
// ---------------------------------------------------------------------------
__global__ void linear_kernel(const float* __restrict__ A,
                              const float* __restrict__ W,
                              const float* __restrict__ bias,
                              int Ktot,
                              float* __restrict__ out,
                              int mode)
{
    __shared__ float sc[2048];
    int tid  = threadIdx.x;
    int wave = tid >> 5;
    int lane = tid & 31;
    int n0   = blockIdx.x << 4;
    bool hi  = lane >= 16;
    int mrow = lane & 15;
    int nn   = n0 + (lane & 15);
    const float* arow = A + (size_t)mrow * Ktot;
    const float* wrow = W + (size_t)nn * Ktot;

    int per_wave = (Ktot >> 5) >> 3;           // chunks per wave (K split 8 ways)
    int kc0 = wave * per_wave;
    v8f acc = {};
    for (int kc = kc0; kc < kc0 + per_wave; ++kc) {
        int ka = (kc << 5) + (hi ? 16 : 0);
        int kb = (kc << 5) + (hi ? 16 : 0);
        v16h af, bf;
#pragma unroll
        for (int j = 0; j < 16; ++j) {
            af[j] = (_Float16)arow[ka + j];
            bf[j] = (_Float16)wrow[kb + j];
        }
        acc = __builtin_amdgcn_wmma_f32_16x16x32_f16(false, af, false, bf, (short)0, acc, false, false);
    }
#pragma unroll
    for (int r = 0; r < 8; ++r) sc[wave * 256 + lane * 8 + r] = acc[r];
    __syncthreads();

    float v = 0.f;
#pragma unroll
    for (int w8 = 0; w8 < 8; ++w8) v += sc[w8 * 256 + tid];
    int lane2 = tid >> 3;
    int r     = tid & 7;
    int row   = r + ((lane2 >= 16) ? 8 : 0);
    int nn2   = n0 + (lane2 & 15);
    float y   = v + bias[nn2];
    if (mode == 0) {
        float e1 = (row < 8) ? eluf(y)  : 0.f;
        float e2 = (row < 8) ? eluf(-y) : 0.f;
        out[row * 4096 + nn2]        = e1;
        out[row * 4096 + nn2 + 2048] = e2;
    } else if (row < 8) {
        float e1 = eluf(y), e2 = eluf(-y);
        out[row * 4096 + nn2]                = e1;
        out[row * 4096 + nn2 + 2048]         = e2;
        out[32768 + row * 4096 + nn2]        = e1;
        out[32768 + row * 4096 + nn2 + 2048] = e2;
    }
}

// ---------------------------------------------------------------------------
extern "C" void kernel_launch(void* const* d_in, const int* in_sizes, int n_in,
                              void* d_out, int out_size, void* d_ws, size_t ws_size,
                              hipStream_t stream)
{
    (void)in_sizes; (void)n_in; (void)out_size; (void)ws_size;
    const float* x       = (const float*)d_in[0];
    const float* m_init1 = (const float*)d_in[1];
    const float* m_und1  = (const float*)d_in[2];
    const float* m_dil1  = (const float*)d_in[3];
    const float* m_init2 = (const float*)d_in[4];
    const float* m_und2  = (const float*)d_in[5];
    const float* m_dil2  = (const float*)d_in[6];
    const float* w1a = (const float*)d_in[7];  const float* b1a = (const float*)d_in[8];
    const float* w2a = (const float*)d_in[9];  const float* b2a = (const float*)d_in[10];
    const float* w3a = (const float*)d_in[11]; const float* b3a = (const float*)d_in[12];
    const float* w1b = (const float*)d_in[13]; const float* b1b = (const float*)d_in[14];
    const float* w2b = (const float*)d_in[15]; const float* b2b = (const float*)d_in[16];
    const float* w3b = (const float*)d_in[17]; const float* b3b = (const float*)d_in[18];
    const float* lw1 = (const float*)d_in[19]; const float* lb1 = (const float*)d_in[20];
    const float* lw2 = (const float*)d_in[21]; const float* lb2 = (const float*)d_in[22];

    char* ws = (char*)d_ws;
    _Float16* h1   = (_Float16*)(ws);                    // 2*32768*192*2 = 25165824 B
    _Float16* h2   = (_Float16*)(ws + 25165824);         // 2*32768*96*2  = 12582912 B
    float*    h3   = (float*)   (ws + 37748736);         // 2*8*8*4096*4  =  2097152 B
    float*    flat = (float*)   (ws + 39845888);         // 16*16384*4    =  1048576 B
    float*    h4   = (float*)   (ws + 40894464);         // 16*4096*4     =   262144 B
    _Float16* w2t  = (_Float16*)(ws + 41156608);         // 165888*2      =   331776 B
    float*    w3t  = (float*)   (ws + 41488384);         // 6912*4        =    27648 B

    prep_kernel<<<675, 256, 0, stream>>>(w2a, w2b, w3a, w3b, w2t, w3t);
    conv1_kernel<<<1024, 64, 0, stream>>>(x, m_init1, m_init2, w1a, b1a, w1b, b1b, h1);
    conv2_kernel<<<4096, 32, 0, stream>>>(h1, m_und1, m_und2, w2t, b2a, b2b, h2);
    conv3_kernel<<<256, 256, 0, stream>>>(h2, m_dil1, m_dil2, w3t, b3a, b3b, h3);
    combine_kernel<<<1024, 256, 0, stream>>>(x, h3, flat);
    linear_kernel<<<128, 256, 0, stream>>>(flat, lw1, lb1, 16384, h4, 0);
    linear_kernel<<<128, 256, 0, stream>>>(h4, lw2, lb2, 4096, (float*)d_out, 1);
}
